// GNNTE_83184926588949
// MI455X (gfx1250) — matrix operand order repaired and checked by reading, not compiled, not run on hardware
//
#include <hip/hip_runtime.h>
#include <hip/hip_bf16.h>

// ---------------------------------------------------------------------------
// GIN forward (3 layers) + mean pooling for gfx1250 (CDNA5, wave32, WMMA).
// GEMMs: bf16 inputs, f32 accumulation via v_wmma_f32_16x16x32_bf16.
// Activations are converted fp32->bf16 once per layer in a streaming pass so
// the WMMA inner loop is pure loads + WMMA (no conversion VALU).
// ---------------------------------------------------------------------------

typedef __attribute__((ext_vector_type(16))) __bf16        v16bf;
typedef __attribute__((ext_vector_type(2)))  __bf16        v2bf;
typedef __attribute__((ext_vector_type(8)))  float         v8f;
typedef __attribute__((ext_vector_type(8)))  unsigned int  v8u;

#define HID 256

// float -> bf16, round-to-nearest-even (finite inputs)
__device__ __forceinline__ unsigned f2bf_u(float f) {
    unsigned u = __float_as_uint(f);
    return (u + 0x7FFFu + ((u >> 16) & 1u)) >> 16;
}
__device__ __forceinline__ unsigned short f2bf(float f) {
    return (unsigned short)f2bf_u(f);
}
__device__ __forceinline__ unsigned pk2(float lo, float hi) {
#if __has_builtin(__builtin_amdgcn_cvt_pk_bf16_f32)
    v2bf r = __builtin_amdgcn_cvt_pk_bf16_f32(lo, hi);
    return __builtin_bit_cast(unsigned, r);
#else
    return f2bf_u(lo) | (f2bf_u(hi) << 16);
#endif
}

// ---------------------------------------------------------------------------
// Weight prep: W[k][n] fp32 (HIDxHID) -> Wt[n][k] bf16, so each WMMA B
// fragment (16 consecutive K for one column n) is a contiguous 32B load.
// ---------------------------------------------------------------------------
__global__ __launch_bounds__(256) void wconv_kernel(const float* __restrict__ W,
                                                    unsigned short* __restrict__ Wt) {
    int t = blockIdx.x * 256 + threadIdx.x;     // t = n*HID + k
    int n = t >> 8;
    int k = t & 255;
    Wt[t] = f2bf(W[k * HID + n]);
}

// ---------------------------------------------------------------------------
// z = h  (float4 copy)
// ---------------------------------------------------------------------------
__global__ __launch_bounds__(256) void copy4_kernel(const float4* __restrict__ src,
                                                    float4* __restrict__ dst, int n4) {
    int i = blockIdx.x * 256 + threadIdx.x;
    if (i < n4) dst[i] = src[i];
}

// ---------------------------------------------------------------------------
// Streaming fp32 -> bf16 convert: 8 floats -> 4 packed uints per thread.
// ---------------------------------------------------------------------------
__global__ __launch_bounds__(256) void cvt_kernel(const float* __restrict__ src,
                                                  uint4* __restrict__ dst, int n8) {
    int i = blockIdx.x * 256 + threadIdx.x;
    if (i >= n8) return;
    const float4* s = (const float4*)(src + (size_t)i * 8);
    float4 a = s[0];
    float4 b = s[1];
    dst[i] = uint4{pk2(a.x, a.y), pk2(a.z, a.w), pk2(b.x, b.y), pk2(b.z, b.w)};
}

// ---------------------------------------------------------------------------
// Edge scatter:  z[dst[e]] += h[src[e]]   (one wave32 per edge, 8 f32/lane,
// native global_atomic_add_f32 via unsafeAtomicAdd)
// ---------------------------------------------------------------------------
__global__ __launch_bounds__(256) void scatter_kernel(const float* __restrict__ h,
                                                      const long long* __restrict__ src,
                                                      const long long* __restrict__ dst,
                                                      float* __restrict__ z, int nEdges) {
    int tid  = blockIdx.x * 256 + threadIdx.x;
    int lane = tid & 31;
    int e    = tid >> 5;
    if (e >= nEdges) return;
    long long s = src[e];
    long long d = dst[e];
    const float* hs = h + (size_t)s * HID + lane * 8;
    float*       zd = z + (size_t)d * HID + lane * 8;
    float4 a = *(const float4*)(hs);
    float4 b = *(const float4*)(hs + 4);
    unsafeAtomicAdd(zd + 0, a.x); unsafeAtomicAdd(zd + 1, a.y);
    unsafeAtomicAdd(zd + 2, a.z); unsafeAtomicAdd(zd + 3, a.w);
    unsafeAtomicAdd(zd + 4, b.x); unsafeAtomicAdd(zd + 5, b.y);
    unsafeAtomicAdd(zd + 6, b.z); unsafeAtomicAdd(zd + 7, b.w);
}

// ---------------------------------------------------------------------------
// GEMM:  out[M x 256] = act(A[M x 256]_bf16 @ W + bias)
// Block = 256 threads = 8 waves; one 16-row M-tile per block (requires
// M % 16 == 0 -- true here: 100000 = 6250*16); each wave owns N-tiles
// {32w, 32w+16}; K-loop 8 x 32 of v_wmma_f32_16x16x32_bf16, fully unrolled.
// Fragment layouts (A/B in a shared contiguous k order, C/D per CDNA5 ISA):
//   A: lane holds row m0 + (lane%16); 16 consecutive K at kb+(lane/16)*16.
//   B: lane holds column nb + (lane%16); 16 consecutive K at kb+(lane/16)*16
//      (contiguous in the n-major transposed weight matrix).
//   C/D: VGPR r -> M = r + 8*(lane/16), N = lane%16.
// ---------------------------------------------------------------------------
template <bool OUT_BF16>
__global__ __launch_bounds__(256) void gin_mlp_gemm(const unsigned short* __restrict__ A_,
                                                    const unsigned short* __restrict__ Wt,
                                                    const float* __restrict__ bias,
                                                    void* __restrict__ out_,
                                                    int relu) {
    const int lane = threadIdx.x & 31;
    const int wave = threadIdx.x >> 5;
    const int half = lane >> 4;   // which 16-lane group
    const int nlo  = lane & 15;
    const int m0   = blockIdx.x << 4;

    const int nb0 = wave << 5;    // two N-tiles: nb0, nb0 + 16
    v8f acc0 = {};
    v8f acc1 = {};

    const unsigned short* Arow = A_ + (size_t)(m0 + nlo) * HID + half * 16;
    const unsigned short* Wr0  = Wt + (size_t)(nb0 + nlo) * HID + half * 16;
    const unsigned short* Wr1  = Wr0 + 16 * HID;

#pragma unroll
    for (int kb = 0; kb < HID; kb += 32) {
        uint4 c0 = *(const uint4*)(Arow + kb);
        uint4 c1 = *(const uint4*)(Arow + kb + 8);
        v16bf a = __builtin_bit_cast(v16bf, v8u{c0.x, c0.y, c0.z, c0.w, c1.x, c1.y, c1.z, c1.w});

        uint4 b00 = *(const uint4*)(Wr0 + kb);
        uint4 b01 = *(const uint4*)(Wr0 + kb + 8);
        uint4 b10 = *(const uint4*)(Wr1 + kb);
        uint4 b11 = *(const uint4*)(Wr1 + kb + 8);
        v16bf b0 = __builtin_bit_cast(v16bf, v8u{b00.x, b00.y, b00.z, b00.w, b01.x, b01.y, b01.z, b01.w});
        v16bf b1 = __builtin_bit_cast(v16bf, v8u{b10.x, b10.y, b10.z, b10.w, b11.x, b11.y, b11.z, b11.w});

        acc0 = __builtin_amdgcn_wmma_f32_16x16x32_bf16(false, a, false, b0, (short)0, acc0, false, false);
        acc1 = __builtin_amdgcn_wmma_f32_16x16x32_bf16(false, a, false, b1, (short)0, acc1, false, false);
    }

    // epilogue (no bounds checks: grid exactly covers M)
#pragma unroll
    for (int t = 0; t < 2; ++t) {
        const int nb = nb0 + (t << 4);
        v8f c = t ? acc1 : acc0;
        const float bv = bias[nb + nlo];
        const size_t base = (size_t)(m0 + (half << 3)) * HID + nb + nlo;
        if (OUT_BF16) {
            unsigned short* o = (unsigned short*)out_ + base;
#pragma unroll
            for (int r = 0; r < 8; ++r) {
                float v = c[r] + bv;
                if (relu) v = fmaxf(v, 0.0f);
                o[(size_t)r * HID] = f2bf(v);
            }
        } else {
            float* o = (float*)out_ + base;
#pragma unroll
            for (int r = 0; r < 8; ++r) {
                float v = c[r] + bv;
                if (relu) v = fmaxf(v, 0.0f);
                o[(size_t)r * HID] = v;
            }
        }
    }
}

// ---------------------------------------------------------------------------
// Pooling
// ---------------------------------------------------------------------------
__global__ __launch_bounds__(256) void zero_kernel(float* __restrict__ p, int n) {
    int i = blockIdx.x * 256 + threadIdx.x;
    if (i < n) p[i] = 0.0f;
}

__global__ __launch_bounds__(256) void pool_kernel(const float* __restrict__ h,
                                                   const long long* __restrict__ batch,
                                                   float* __restrict__ out,
                                                   float* __restrict__ counts, int N) {
    int tid  = blockIdx.x * 256 + threadIdx.x;
    int node = tid >> 6;          // 64 threads per node, 4 floats each
    int q    = tid & 63;
    if (node >= N) return;
    int g = (int)batch[node];
    const float* hr = h + (size_t)node * HID + q * 4;
    float*       o  = out + (size_t)g * HID + q * 4;
    float4 v = *(const float4*)hr;
    unsafeAtomicAdd(o + 0, v.x); unsafeAtomicAdd(o + 1, v.y);
    unsafeAtomicAdd(o + 2, v.z); unsafeAtomicAdd(o + 3, v.w);
    if (q == 0) unsafeAtomicAdd(&counts[g], 1.0f);
}

__global__ __launch_bounds__(256) void norm_kernel(float* __restrict__ out,
                                                   const float* __restrict__ counts, int n) {
    int i = blockIdx.x * 256 + threadIdx.x;
    if (i < n) out[i] /= fmaxf(counts[i >> 8], 1.0f);
}

// ---------------------------------------------------------------------------
// Host driver
// ---------------------------------------------------------------------------
extern "C" void kernel_launch(void* const* d_in, const int* in_sizes, int n_in,
                              void* d_out, int out_size, void* d_ws, size_t ws_size,
                              hipStream_t stream) {
    const float*     x     = (const float*)d_in[0];
    const long long* eidx  = (const long long*)d_in[1];
    const long long* batch = (const long long*)d_in[2];
    const float*     W1    = (const float*)d_in[3];
    const float*     B1    = (const float*)d_in[4];
    const float*     W2    = (const float*)d_in[5];
    const float*     B2    = (const float*)d_in[6];
    float*           out   = (float*)d_out;

    const int N       = in_sizes[0] / HID;      // 100000 (multiple of 16)
    const int nEdges  = in_sizes[1] / 2;        // 320000
    const int nLayers = in_sizes[3] / (HID * HID);
    const int G       = out_size / HID;         // 1024

    const long long* src = eidx;
    const long long* dst = eidx + nEdges;

    // workspace layout
    const size_t NH = (size_t)N * HID;
    float*          z      = (float*)d_ws;                      // fp32  N*HID
    float*          hbuf   = z + NH;                            // fp32  N*HID
    unsigned short* zbf    = (unsigned short*)(hbuf + NH);      // bf16  N*HID
    unsigned short* ybf    = zbf + NH;                          // bf16  N*HID
    unsigned short* w1t    = ybf + NH;                          // bf16  HID*HID
    unsigned short* w2t    = w1t + HID * HID;                   // bf16  HID*HID
    float*          counts = (float*)(w2t + HID * HID);         // fp32  G
    (void)ws_size; (void)n_in;

    const int gemmBlocks = N / 16;              // exact
    const int copyN4     = (int)(NH / 4);
    const int copyBlocks = (copyN4 + 255) / 256;
    const int cvtN8      = (int)(NH / 8);
    const int cvtBlocks  = (cvtN8 + 255) / 256;
    const int scatBlocks = (nEdges * 32 + 255) / 256;
    const int wBlocks    = (HID * HID) / 256;

    for (int i = 0; i < nLayers; ++i) {
        const float* hin = (i == 0) ? x : hbuf;

        // weight transpose+convert for this layer
        wconv_kernel<<<wBlocks, 256, 0, stream>>>(W1 + (size_t)i * HID * HID, w1t);
        wconv_kernel<<<wBlocks, 256, 0, stream>>>(W2 + (size_t)i * HID * HID, w2t);

        // z = (1 + eps) * h + sum_{j in N(i)} h_j  (eps = 0)
        copy4_kernel<<<copyBlocks, 256, 0, stream>>>((const float4*)hin, (float4*)z, copyN4);
        scatter_kernel<<<scatBlocks, 256, 0, stream>>>(hin, src, dst, z, nEdges);

        // zbf = bf16(z)  (single streaming conversion pass)
        cvt_kernel<<<cvtBlocks, 256, 0, stream>>>(z, (uint4*)zbf, cvtN8);

        // MLP: ybf = relu(z @ W1 + b1) [bf16];  h = (relu?)(ybf @ W2 + b2) [f32]
        gin_mlp_gemm<true><<<gemmBlocks, 256, 0, stream>>>(
            zbf, w1t, B1 + (size_t)i * HID, (void*)ybf, 1);
        gin_mlp_gemm<false><<<gemmBlocks, 256, 0, stream>>>(
            ybf, w2t, B2 + (size_t)i * HID, (void*)hbuf,
            (i < nLayers - 1) ? 1 : 0);
    }

    // mean pooling over sorted batch ids
    zero_kernel<<<(G * HID + 255) / 256, 256, 0, stream>>>(out, G * HID);
    zero_kernel<<<(G + 255) / 256, 256, 0, stream>>>(counts, G);
    pool_kernel<<<(N * 64 + 255) / 256, 256, 0, stream>>>(hbuf, batch, out, counts, N);
    norm_kernel<<<(G * HID + 255) / 256, 256, 0, stream>>>(out, counts, G * HID);
}
